// LateralAttention_5755256177419
// MI455X (gfx1250) — matrix operand order repaired and checked by reading, not compiled, not run on hardware
//
#include <hip/hip_runtime.h>

// ---------------------------------------------------------------------------
// LateralAttention for MI455X (gfx1250, wave32, WMMA, async global->LDS).
//   B=32, S=577, HID=768, NH=12, HD=64.  S padded to SP=640 in workspace.
//   Kernel 1: fused QKV projection (bf16 WMMA, fp32 accum) + bias + lateral
//             mix + 1/sqrt(HD) folded into Q.  Q,K stored [B][NH][SP][HD],
//             V stored pre-transposed [B][NH][HD][SP] (LDS tile transpose).
//   Kernel 2: flash attention; K/V blocks double-buffered in LDS via
//             global_load_async_to_lds_b128 + s_wait_asynccnt.
// ---------------------------------------------------------------------------

#define B_   32
#define S_   577
#define SP_  640      // padded sequence (multiple of 64)
#define HID_ 768
#define NH_  12
#define HD_  64

typedef __attribute__((ext_vector_type(16))) __bf16 v16bf;
typedef __attribute__((ext_vector_type(2)))  __bf16 v2bf;
typedef __attribute__((ext_vector_type(8)))  float  v8f;

union Frag { v16bf v; unsigned int u[8]; };

__device__ __forceinline__ unsigned pk2bf(float lo, float hi) {
#if __has_builtin(__builtin_amdgcn_cvt_pk_bf16_f32)
  union { v2bf h; unsigned u; } r;
  r.h = __builtin_amdgcn_cvt_pk_bf16_f32(lo, hi);
  return r.u;
#else
  union { v2bf h; unsigned u; } r;
  r.h = v2bf{(__bf16)lo, (__bf16)hi};
  return r.u;
#endif
}
__device__ __forceinline__ unsigned short bf1(float f) {
  union { __bf16 h; unsigned short u; } r;
  r.h = (__bf16)f;
  return r.u;
}
__device__ __forceinline__ v8f zero8() {
  v8f z = {0.f, 0.f, 0.f, 0.f, 0.f, 0.f, 0.f, 0.f};
  return z;
}

// Async 16-byte global -> LDS copy (CDNA5 GLOBAL_LOAD_ASYNC_TO_LDS_B128,
// tracked by ASYNCcnt).  lds_off is a byte offset in LDS space.
__device__ __forceinline__ void async_b128(unsigned lds_off, const void* g) {
  unsigned long long ga = (unsigned long long)g;
  asm volatile("global_load_async_to_lds_b128 %0, %1, off"
               :: "v"(lds_off), "v"(ga) : "memory");
}
__device__ __forceinline__ void wait_async4() {
  asm volatile("s_wait_asynccnt 0x4" ::: "memory");
}
__device__ __forceinline__ void wait_async0() {
  asm volatile("s_wait_asynccnt 0x0" ::: "memory");
}

// ---------------------------------------------------------------------------
// Kernel 1: Y = X @ W^T + b  (+ lateral mix, + 0.125 scale for Q)
// grid = (B*SP/64, HID/64, 3[q,k,v]), block = 128 (4 waves)
// ---------------------------------------------------------------------------
__global__ __launch_bounds__(128) void qkv_proj_kernel(
    const float* __restrict__ X,  const float* __restrict__ lat,
    const float* __restrict__ Wq, const float* __restrict__ bq,
    const float* __restrict__ Wk, const float* __restrict__ bk,
    const float* __restrict__ Wv, const float* __restrict__ bv,
    const int* __restrict__ mixp, unsigned short* __restrict__ ws)
{
  __shared__ __align__(16) unsigned short Xs[64][40];    // 64 x 32 bf16, rows 80B
  __shared__ __align__(16) unsigned short Wsh[64][40];
  __shared__ __align__(16) unsigned short Vls[64][72];   // V transpose tile [d][s]

  const int tid  = threadIdx.x;
  const int lane = tid & 31;
  const int wave = tid >> 5;
  const int z    = blockIdx.z;                // 0=q, 1=k, 2=v
  const int n0   = blockIdx.y * 64;           // output-feature tile (one head)
  const int m0   = blockIdx.x * 64;
  const int b    = m0 / SP_;                  // SP_ % 64 == 0 -> uniform per WG
  const int s0   = m0 % SP_;
  const int mix  = mixp[0];

  const float* Wp = (z == 0) ? Wq : (z == 1) ? Wk : Wv;
  const float* bp = (z == 0) ? bq : (z == 1) ? bk : bv;

  v8f acc[4];
  #pragma unroll
  for (int c = 0; c < 4; ++c) acc[c] = zero8();

  for (int kt = 0; kt < HID_ / 32; ++kt) {
    const int k0 = kt * 32;
    // Phase 1: branchless clamped loads, all 8 b128 loads in flight together.
    float4 xv[4], wv[4];
    #pragma unroll
    for (int i = 0; i < 4; ++i) {
      int e = tid + i * 128;
      int row = e >> 3, c4 = e & 7;           // 8 float4 per 32-wide row
      int s = s0 + row;
      int sc = (s < S_) ? s : (S_ - 1);       // clamp; zero-select below
      xv[i] = *(const float4*)(X + (size_t)(b * S_ + sc) * HID_ + k0 + c4 * 4);
      wv[i] = *(const float4*)(Wp + (size_t)(n0 + row) * HID_ + k0 + c4 * 4);
    }
    // Phase 2: convert + packed bf16 stores (ds_store_b64 each)
    #pragma unroll
    for (int i = 0; i < 4; ++i) {
      int e = tid + i * 128;
      int row = e >> 3, c4 = e & 7;
      float4 x4 = xv[i];
      if (s0 + row >= S_) { x4.x = 0.f; x4.y = 0.f; x4.z = 0.f; x4.w = 0.f; }
      *(uint2*)&Xs[row][c4 * 4]  = make_uint2(pk2bf(x4.x, x4.y), pk2bf(x4.z, x4.w));
      *(uint2*)&Wsh[row][c4 * 4] = make_uint2(pk2bf(wv[i].x, wv[i].y), pk2bf(wv[i].z, wv[i].w));
    }
    __syncthreads();

    // A fragment: 16x32 bf16 (this wave's 16 rows)
    Frag a;
    {
      const unsigned int* Xrow = (const unsigned int*)(&Xs[wave * 16 + (lane & 15)][0]);
      const int base = (lane < 16) ? 0 : 8;
      #pragma unroll
      for (int i = 0; i < 4; ++i) { a.u[i] = Xrow[base + i]; a.u[4 + i] = Xrow[4 + base + i]; }
    }
    // B fragments: 32x16 bf16, one per 16-wide output-feature chunk
    #pragma unroll
    for (int c = 0; c < 4; ++c) {
      Frag bfr;
      const unsigned int* Wrow = (const unsigned int*)(&Wsh[c * 16 + (lane & 15)][0]);
      const int base = (lane < 16) ? 0 : 8;
      #pragma unroll
      for (int v = 0; v < 8; ++v) bfr.u[v] = Wrow[base + v];
      acc[c] = __builtin_amdgcn_wmma_f32_16x16x32_bf16(
          false, a.v, false, bfr.v, (short)0, acc[c], false, false);
    }
    __syncthreads();
  }

  // Epilogue: bias + lateral mix + Q scale (branchless, batched lat loads).
  const size_t QSZ = (size_t)B_ * NH_ * SP_ * HD_;
  const int hh = n0 >> 6;                       // this tile's head
  const int rowbase = wave * 16 + ((lane & 16) ? 8 : 0);
  const bool domix = (z == mix);                // uniform across block

  float latv[4][8];
  if (domix) {
    const float* latb = lat + (size_t)(b * NH_ + hh) * S_ * HD_;
    #pragma unroll
    for (int c = 0; c < 4; ++c) {
      const int d = c * 16 + (lane & 15);
      #pragma unroll
      for (int r = 0; r < 8; ++r) {
        int s = s0 + rowbase + r;
        int sc = (s < S_) ? s : (S_ - 1);       // clamped, non-divergent load
        latv[c][r] = latb[(size_t)sc * HD_ + d];
      }
    }
  }

  const float qscale = (z == 0) ? 0.125f : 1.0f;  // 1/sqrt(HD) folded into Q
  float vals[4][8];
  #pragma unroll
  for (int c = 0; c < 4; ++c) {
    const float bias = bp[n0 + c * 16 + (lane & 15)];
    #pragma unroll
    for (int r = 0; r < 8; ++r) {
      int s = s0 + rowbase + r;
      float val = acc[c][r] + bias;
      if (domix) val *= latv[c][r];
      val *= qscale;
      vals[c][r] = (s < S_) ? val : 0.0f;       // pad rows -> exact zero
    }
  }

  if (z != 2) {
    // Q,K: [B][NH][SP][HD] bf16, coalesced u16 stores along d
    #pragma unroll
    for (int c = 0; c < 4; ++c) {
      const int d = c * 16 + (lane & 15);
      #pragma unroll
      for (int r = 0; r < 8; ++r) {
        int s = s0 + rowbase + r;
        ws[(size_t)z * QSZ + ((size_t)(b * NH_ + hh) * SP_ + s) * HD_ + d] = bf1(vals[c][r]);
      }
    }
  } else {
    // V: transpose tile in LDS, then coalesced rows of [B][NH][HD][SP]
    #pragma unroll
    for (int c = 0; c < 4; ++c) {
      const int d = c * 16 + (lane & 15);
      uint4 u4;
      u4.x = pk2bf(vals[c][0], vals[c][1]);
      u4.y = pk2bf(vals[c][2], vals[c][3]);
      u4.z = pk2bf(vals[c][4], vals[c][5]);
      u4.w = pk2bf(vals[c][6], vals[c][7]);
      *(uint4*)&Vls[d][rowbase] = u4;          // 8 consecutive s, 16B aligned
    }
    __syncthreads();
    unsigned short* wsv = ws + 2 * QSZ + (size_t)(b * NH_ + hh) * HD_ * SP_;
    #pragma unroll
    for (int i = 0; i < 4; ++i) {
      int e = tid + i * 128;
      int d = e >> 3, c4 = e & 7;
      uint4 u4 = *(const uint4*)&Vls[d][c4 * 8];
      *(uint4*)(wsv + (size_t)d * SP_ + s0 + c4 * 8) = u4;
    }
  }
}

// ---------------------------------------------------------------------------
// Kernel 2: flash attention.
// grid = (SP/64, NH, B), block = 128 (4 waves, each owns 16 query rows)
// ---------------------------------------------------------------------------
#define KBUF 4608u   // 32*72*2
#define VBUF 5120u   // 64*40*2

__global__ __launch_bounds__(128) void lateral_attn_kernel(
    const unsigned short* __restrict__ ws, float* __restrict__ out)
{
  __shared__ __align__(16) unsigned short Kt[2][32][72];   // [key][dim], rows 144B
  __shared__ __align__(16) unsigned short Vt[2][64][40];   // [dim][key], rows 80B
  __shared__ __align__(16) unsigned short Qs[4][16][72];   // per-wave Q tile
  __shared__ __align__(16) unsigned short Ps[4][16][40];   // per-wave P tile

  const int tid = threadIdx.x, lane = tid & 31, wave = tid >> 5;
  const int b = blockIdx.z, h = blockIdx.y;
  const int q0 = (blockIdx.x * 4 + wave) * 16;

  const size_t QSZ = (size_t)B_ * NH_ * SP_ * HD_;
  const unsigned short* Qp = ws + (size_t)(b * NH_ + h) * SP_ * HD_;
  const unsigned short* Kp = ws + QSZ + (size_t)(b * NH_ + h) * SP_ * HD_;
  const unsigned short* Vp = ws + 2 * QSZ + (size_t)(b * NH_ + h) * HD_ * SP_;

  const unsigned ldsK = (unsigned)(size_t)(void*)&Kt[0][0][0];
  const unsigned ldsV = (unsigned)(size_t)(void*)&Vt[0][0][0];
  const unsigned ldsQ = (unsigned)(size_t)(void*)&Qs[wave][0][0];

  // K/V block staging: 4 async b128 instructions per wave per block
  auto issue_kv = [&](int jb, int buf) {
    const int j0 = jb * 32;
    #pragma unroll
    for (int i = 0; i < 2; ++i) {
      int e = tid + i * 128;
      int key = e >> 3, kc4 = e & 7;            // K: 32 rows x 8 chunks
      async_b128(ldsK + (unsigned)buf * KBUF + key * 144u + kc4 * 16u,
                 Kp + (size_t)(j0 + key) * HD_ + kc4 * 8);
      int d = e >> 2, vc4 = e & 3;              // V: 64 rows x 4 chunks
      async_b128(ldsV + (unsigned)buf * VBUF + d * 80u + vc4 * 16u,
                 Vp + (size_t)d * SP_ + j0 + vc4 * 8);
    }
  };

  // Prologue: async-stage Q tile + first K/V block
  #pragma unroll
  for (int i = 0; i < 4; ++i) {
    int e = lane + i * 32;
    int row = e >> 3, c4 = e & 7;
    async_b128(ldsQ + row * 144u + c4 * 16u, Qp + (size_t)(q0 + row) * HD_ + c4 * 8);
  }
  issue_kv(0, 0);
  wait_async0();
  __syncthreads();

  // Q A-fragments (head-dim chunks 0..31, 32..63)
  Frag qa[2];
  {
    const unsigned int* Qrow = (const unsigned int*)(&Qs[wave][lane & 15][0]);
    const int base = (lane < 16) ? 0 : 8;
    #pragma unroll
    for (int dc = 0; dc < 2; ++dc)
      #pragma unroll
      for (int i = 0; i < 4; ++i) {
        qa[dc].u[i]     = Qrow[dc * 16 + base + i];
        qa[dc].u[4 + i] = Qrow[dc * 16 + 4 + base + i];
      }
  }

  v8f acc[4];
  #pragma unroll
  for (int c = 0; c < 4; ++c) acc[c] = zero8();
  float m8[8], l8[8];
  #pragma unroll
  for (int r = 0; r < 8; ++r) { m8[r] = -3.0e38f; l8[r] = 0.0f; }

  const int NBLK = (S_ + 31) / 32;   // 19
  for (int jb = 0; jb < NBLK; ++jb) {
    const int cur = jb & 1;
    const int j0 = jb * 32;
    if (jb + 1 < NBLK) {             // double-buffer: prefetch next block
      issue_kv(jb + 1, cur ^ 1);
      wait_async4();                 // current block's 4 asyncs retired
    } else {
      wait_async0();
    }
    __syncthreads();

    // Scores: two 16x16 C tiles (key chunks), K=64 over head dim.
    v8f sc[2];
    #pragma unroll
    for (int kc = 0; kc < 2; ++kc) {
      sc[kc] = zero8();
      #pragma unroll
      for (int dc = 0; dc < 2; ++dc) {
        Frag kb;
        const unsigned int* Krow = (const unsigned int*)(&Kt[cur][kc * 16 + (lane & 15)][0]);
        const int base = (lane < 16) ? 0 : 8;
        #pragma unroll
        for (int v = 0; v < 8; ++v) kb.u[v] = Krow[dc * 16 + base + v];
        sc[kc] = __builtin_amdgcn_wmma_f32_16x16x32_bf16(
            false, qa[dc].v, false, kb.v, (short)0, sc[kc], false, false);
      }
      if (j0 + kc * 16 + (lane & 15) >= S_) {   // mask padded keys
        #pragma unroll
        for (int r = 0; r < 8; ++r) sc[kc][r] = -1.0e30f;
      }
    }

    // Online softmax (row reductions over the 16 lanes of each half-wave)
    #pragma unroll
    for (int r = 0; r < 8; ++r) {
      float mx = fmaxf(sc[0][r], sc[1][r]);
      #pragma unroll
      for (int off = 1; off < 16; off <<= 1) mx = fmaxf(mx, __shfl_xor(mx, off, 16));
      float mn = fmaxf(m8[r], mx);
      float alpha = __expf(m8[r] - mn);
      float p0 = __expf(sc[0][r] - mn);
      float p1 = __expf(sc[1][r] - mn);
      sc[0][r] = p0; sc[1][r] = p1;
      float rs = p0 + p1;
      #pragma unroll
      for (int off = 1; off < 16; off <<= 1) rs += __shfl_xor(rs, off, 16);
      l8[r] = alpha * l8[r] + rs;
      m8[r] = mn;
      #pragma unroll
      for (int c = 0; c < 4; ++c) acc[c][r] *= alpha;
    }

    // P: C-layout -> LDS [16][32] bf16 -> A-fragment
    {
      const int rowb = (lane & 16) ? 8 : 0;
      const int col  = lane & 15;
      #pragma unroll
      for (int r = 0; r < 8; ++r) {
        Ps[wave][rowb + r][col]      = bf1(sc[0][r]);
        Ps[wave][rowb + r][col + 16] = bf1(sc[1][r]);
      }
    }
    Frag pa;
    {
      const unsigned int* Prow = (const unsigned int*)(&Ps[wave][lane & 15][0]);
      const int base = (lane < 16) ? 0 : 8;
      #pragma unroll
      for (int i = 0; i < 4; ++i) { pa.u[i] = Prow[base + i]; pa.u[4 + i] = Prow[4 + base + i]; }
    }

    // Context: acc[c] += P(16x32) x V(32x16) per 16-wide head-dim chunk
    #pragma unroll
    for (int c = 0; c < 4; ++c) {
      Frag vb;
      const unsigned int* Vrow = (const unsigned int*)(&Vt[cur][c * 16 + (lane & 15)][0]);
      const int base = (lane < 16) ? 0 : 8;
      #pragma unroll
      for (int v = 0; v < 8; ++v) vb.u[v] = Vrow[base + v];
      acc[c] = __builtin_amdgcn_wmma_f32_16x16x32_bf16(
          false, pa.v, false, vb.v, (short)0, acc[c], false, false);
    }
    __syncthreads();
  }

  // Normalize + scatter to [B, S, HID] fp32 output
  const int rowb = (lane & 16) ? 8 : 0;
  #pragma unroll
  for (int r = 0; r < 8; ++r) {
    int s = q0 + rowb + r;
    if (s < S_) {
      float inv = 1.0f / l8[r];
      #pragma unroll
      for (int c = 0; c < 4; ++c)
        out[(size_t)(b * S_ + s) * HID_ + h * HD_ + c * 16 + (lane & 15)] = acc[c][r] * inv;
    }
  }
}

// ---------------------------------------------------------------------------
extern "C" void kernel_launch(void* const* d_in, const int* in_sizes, int n_in,
                              void* d_out, int out_size, void* d_ws, size_t ws_size,
                              hipStream_t stream) {
  (void)in_sizes; (void)n_in; (void)out_size; (void)ws_size;
  const float* X   = (const float*)d_in[0];
  const float* lat = (const float*)d_in[1];
  const float* Wq  = (const float*)d_in[2];
  const float* bq  = (const float*)d_in[3];
  const float* Wk  = (const float*)d_in[4];
  const float* bk  = (const float*)d_in[5];
  const float* Wv  = (const float*)d_in[6];
  const float* bv  = (const float*)d_in[7];
  const int*   mix = (const int*)d_in[8];
  unsigned short* ws = (unsigned short*)d_ws;   // 3 * B*NH*SP*HD bf16 (~94 MB)
  float* out = (float*)d_out;

  dim3 pgrid(B_ * SP_ / 64, HID_ / 64, 3);
  qkv_proj_kernel<<<pgrid, 128, 0, stream>>>(X, lat, Wq, bq, Wk, bk, Wv, bv, mix, ws);

  dim3 agrid(SP_ / 64, NH_, B_);
  lateral_attn_kernel<<<agrid, 128, 0, stream>>>(ws, out);
}
